// TopoGNN_49409303773498
// MI455X (gfx1250) — matrix operand order repaired and checked by reading, not compiled, not run on hardware
//
#include <hip/hip_runtime.h>
#include <hip/hip_bf16.h>

// ---------------------------------------------------------------------------
// TopoGNN fused InteractionNetwork step for gfx1250 (MI455X), wave32 + WMMA.
//   h_n  = swish(nodes @ W_emb_n + b)          (k_node_embed, bf16 out -> ws)
//   new_e= swish([h_e|h_n[r]|h_n[s]] @ W_e + b) + atomic scatter-add (k_edge)
//   new_n= swish([agg|h_n] @ W_n + b); out=[nodes0|new_n]          (k_node)
// All GEMM tiles use v_wmma_f32_16x16x32_bf16 (fp32 accumulate).
// ---------------------------------------------------------------------------

typedef __bf16 bf16_t;
typedef __attribute__((ext_vector_type(16))) __bf16          v16bf;
typedef __attribute__((ext_vector_type(8)))  float           v8f;
typedef __attribute__((ext_vector_type(8)))  unsigned short  v8us;

union BfPack {
    v16bf          bf;
    v8us           us2[2];
    unsigned short s[16];
};

__device__ __forceinline__ unsigned short to_bf16_bits(float f) {
    // native fptrunc -> v_cvt_pk_bf16_f32 on gfx1250
    return __builtin_bit_cast(unsigned short, (bf16_t)f);
}

__device__ __forceinline__ float swishf(float x) {
    // x * sigmoid(x); v_exp_f32 + v_rcp_f32 (avoid IEEE div expansion)
    return x * __builtin_amdgcn_rcpf(1.0f + __expf(-x));
}

// Pack a KxN(=64) fp32 weight into LDS as per-lane bf16 B-operands.
// B layout (32x16 bf16 per chunk c, per N-tile t): lane L holds
//   element i -> B[k][n], k = c*32 + (L>>4)*8 + (i&7) + (i>>3)*16, n = t*16 + (L&15).
// Flat LDS index = ((c*4 + t)*32 + lane)*16 + i.
__device__ __forceinline__ void pack_B(const float* __restrict__ W,
                                       unsigned short* lds,
                                       int chunks, int realK,
                                       int tid, int nthreads) {
    int total = chunks * 4 * 32 * 16;
    for (int idx = tid; idx < total; idx += nthreads) {
        int i    = idx & 15;
        int lane = (idx >> 4) & 31;
        int ct   = idx >> 9;
        int t    = ct & 3;
        int c    = ct >> 2;
        int k    = c * 32 + (lane >> 4) * 8 + (i & 7) + (i >> 3) * 16;
        int n    = t * 16 + (lane & 15);
        float v  = (k < realK) ? W[(size_t)k * 64 + n] : 0.0f;
        lds[idx] = to_bf16_bits(v);
    }
}

__device__ __forceinline__ v16bf load_B_lds(const unsigned short* lds,
                                            int c, int t, int lane) {
    return *(const v16bf*)(lds + ((((c * 4 + t) * 32) + lane) << 4));
}

// ---------------------------------------------------------------------------
__global__ void __launch_bounds__(256) k_zero(float* __restrict__ agg, int n) {
    int i = blockIdx.x * blockDim.x + threadIdx.x;
    if (i < n) agg[i] = 0.0f;
}

// ---------------------------------------------------------------------------
// h_n = swish(nodes[N,32] @ W_emb_n[32,64] + b), stored bf16 [N,64].
__global__ void __launch_bounds__(256) k_node_embed(
        const float* __restrict__ nodes,
        const float* __restrict__ W,
        const float* __restrict__ bias,
        unsigned short* __restrict__ h_n_bf,
        int N) {
    __shared__ __attribute__((aligned(32))) unsigned short sB[1 * 4 * 32 * 16];
    pack_B(W, sB, /*chunks=*/1, /*realK=*/32, threadIdx.x, blockDim.x);
    __syncthreads();

    const int lane   = threadIdx.x & 31;
    const int r      = lane & 15;
    const int half   = lane >> 4;
    const int wid    = (blockIdx.x * blockDim.x + threadIdx.x) >> 5;
    const int nWaves = (gridDim.x * blockDim.x) >> 5;
    const int nTiles = N >> 4;

    float bv[4];
#pragma unroll
    for (int t = 0; t < 4; ++t) bv[t] = bias[t * 16 + r];

    // weight B-operands are reused every tile: keep in registers (4 x 8 VGPR)
    v16bf Breg[4];
#pragma unroll
    for (int t = 0; t < 4; ++t) Breg[t] = load_B_lds(sB, 0, t, lane);

    for (int tile = wid; tile < nTiles; tile += nWaves) {
        const int m0 = tile << 4;
        // A tile: 32-bit base offset, constant sub-offsets (fold to IOFFSET)
        const unsigned aOff = (unsigned)(m0 + r) * 32u + (unsigned)half * 8u;
        BfPack a;
#pragma unroll
        for (int i = 0; i < 8; ++i) {
            a.s[i]     = to_bf16_bits(nodes[aOff + i]);        // K = half*8+i
            a.s[i + 8] = to_bf16_bits(nodes[aOff + 16 + i]);   // K = 16+half*8+i
        }
        // D-store base: element (i,t) lands at base + i*64 + t*16
        const unsigned dOff = (unsigned)(m0 + 8 * half) * 64u + (unsigned)r;
#pragma unroll
        for (int t = 0; t < 4; ++t) {
            v8f c = {};
            c = __builtin_amdgcn_wmma_f32_16x16x32_bf16(false, a.bf, false, Breg[t],
                                                        (short)0, c, false, false);
#pragma unroll
            for (int i = 0; i < 8; ++i) {
                float y = swishf(c[i] + bv[t]);
                h_n_bf[dOff + (unsigned)(i * 64 + t * 16)] = to_bf16_bits(y);
            }
        }
    }
}

// ---------------------------------------------------------------------------
// Fused edge block: h_e = swish(edges@W_emb_e+b); new_e = swish(e_in@W_e+b);
// agg[recv] += new_e.   16 edges per wave-tile; 28 WMMAs per tile.
#define WAVES_PER_BLOCK 8
__global__ void __launch_bounds__(256) k_edge(
        const float* __restrict__ edges,
        const int*   __restrict__ senders,
        const int*   __restrict__ receivers,
        const float* __restrict__ W_emb_e,
        const float* __restrict__ b_emb_e,
        const float* __restrict__ W_e,
        const float* __restrict__ b_e,
        const unsigned short* __restrict__ h_n_bf,
        float* __restrict__ agg,
        int E) {
    __shared__ __attribute__((aligned(32))) unsigned short sBemb[1 * 4 * 32 * 16];  //  4 KB
    __shared__ __attribute__((aligned(32))) unsigned short sBe  [6 * 4 * 32 * 16];  // 24 KB
    __shared__ __attribute__((aligned(32))) unsigned short sHe  [WAVES_PER_BLOCK][16 * 64]; // 16 KB
    pack_B(W_emb_e, sBemb, 1, 16,  threadIdx.x, blockDim.x);   // K padded 16->32
    pack_B(W_e,     sBe,   6, 192, threadIdx.x, blockDim.x);
    __syncthreads();

    const int lane   = threadIdx.x & 31;
    const int r      = lane & 15;
    const int half   = lane >> 4;
    const int wslot  = threadIdx.x >> 5;
    const int wid    = (blockIdx.x * blockDim.x + threadIdx.x) >> 5;
    const int nWaves = (gridDim.x * blockDim.x) >> 5;
    const int nTiles = E >> 4;
    unsigned short* he = &sHe[wslot][0];

    float bE[4], bEmb[4];
#pragma unroll
    for (int t = 0; t < 4; ++t) { bE[t] = b_e[t * 16 + r]; bEmb[t] = b_emb_e[t * 16 + r]; }

    // embedding weights are tiny and reused every tile: keep in registers
    v16bf BembReg[4];
#pragma unroll
    for (int t = 0; t < 4; ++t) BembReg[t] = load_B_lds(sBemb, 0, t, lane);

    for (int tile = wid; tile < nTiles; tile += nWaves) {
        const int m0 = tile << 4;

        // ---- edge embedding: A = raw edge feats, K real 16, padded to 32 ----
        const unsigned eOff = (unsigned)(m0 + r) * 16u + (unsigned)half * 8u;
        BfPack a0;
#pragma unroll
        for (int i = 0; i < 8; ++i) {
            a0.s[i]     = to_bf16_bits(edges[eOff + i]);
            a0.s[i + 8] = 0;                                   // K>=16 padding
        }
#pragma unroll
        for (int t = 0; t < 4; ++t) {
            v8f c = {};
            c = __builtin_amdgcn_wmma_f32_16x16x32_bf16(false, a0.bf, false, BembReg[t],
                                                        (short)0, c, false, false);
#pragma unroll
            for (int i = 0; i < 8; ++i) {
                float y = swishf(c[i] + bEmb[t]);
                // D layout -> row-major LDS stage (cross-lane transpose point)
                he[(i + 8 * half) * 64 + t * 16 + r] = to_bf16_bits(y);
            }
        }
        // make wave-local LDS stage visible across lanes before A-layout reads
        asm volatile("s_wait_dscnt 0" ::: "memory");

        // ---- build A chunks (K = 192 = 6 x 32) ----
        const int recvR = receivers[m0 + r];
        const int sendR = senders[m0 + r];
        BfPack A[6];
        const unsigned short* hrow = he + r * 64;
        const unsigned rOff = (unsigned)recvR * 64u + (unsigned)half * 8u;
        const unsigned sOff = (unsigned)sendR * 64u + (unsigned)half * 8u;
#pragma unroll
        for (int c = 0; c < 2; ++c) {
            A[c].us2[0]     = *(const v8us*)(hrow + c * 32 + half * 8);
            A[c].us2[1]     = *(const v8us*)(hrow + c * 32 + 16 + half * 8);
            A[2 + c].us2[0] = *(const v8us*)(h_n_bf + rOff + (unsigned)(c * 32));
            A[2 + c].us2[1] = *(const v8us*)(h_n_bf + rOff + (unsigned)(c * 32 + 16));
            A[4 + c].us2[0] = *(const v8us*)(h_n_bf + sOff + (unsigned)(c * 32));
            A[4 + c].us2[1] = *(const v8us*)(h_n_bf + sOff + (unsigned)(c * 32 + 16));
        }

        // scatter base offset per accumulator row M = i + 8*half (t*16 folds)
        unsigned aggOff[8];
#pragma unroll
        for (int i = 0; i < 8; ++i) {
            int dst = __shfl(recvR, i + 8 * half, 32);   // receivers[m0+i+8*half]
            aggOff[i] = (unsigned)dst * 64u + (unsigned)r;
        }

        // ---- main GEMM [16,192]@[192,64]: one live acc tile at a time ----
#pragma unroll
        for (int t = 0; t < 4; ++t) {
            v8f c = {};
#pragma unroll
            for (int cc = 0; cc < 6; ++cc) {
                v16bf b = load_B_lds(sBe, cc, t, lane);
                c = __builtin_amdgcn_wmma_f32_16x16x32_bf16(false, A[cc].bf, false, b,
                                                            (short)0, c, false, false);
            }
            // swish + scatter-add (segment_sum over receivers), fused epilogue
#pragma unroll
            for (int i = 0; i < 8; ++i) {
                float y = swishf(c[i] + bE[t]);
                __hip_atomic_fetch_add(agg + aggOff[i] + (unsigned)(t * 16), y,
                                       __ATOMIC_RELAXED, __HIP_MEMORY_SCOPE_AGENT);
            }
        }
    }
}

// ---------------------------------------------------------------------------
// new_n = swish([agg|h_n] @ W_n[128,64] + b); out = [nodes0 | new_n]  [N,96]
__global__ void __launch_bounds__(256) k_node(
        const float* __restrict__ nodes,
        const float* __restrict__ W_n,
        const float* __restrict__ b_n,
        const float* __restrict__ agg,
        const unsigned short* __restrict__ h_n_bf,
        float* __restrict__ out,
        int N) {
    __shared__ __attribute__((aligned(32))) unsigned short sBn[4 * 4 * 32 * 16]; // 16 KB
    pack_B(W_n, sBn, 4, 128, threadIdx.x, blockDim.x);
    __syncthreads();

    const int lane   = threadIdx.x & 31;
    const int r      = lane & 15;
    const int half   = lane >> 4;
    const int wid    = (blockIdx.x * blockDim.x + threadIdx.x) >> 5;
    const int nWaves = (gridDim.x * blockDim.x) >> 5;
    const int nTiles = N >> 4;

    float bv[4];
#pragma unroll
    for (int t = 0; t < 4; ++t) bv[t] = b_n[t * 16 + r];

    for (int tile = wid; tile < nTiles; tile += nWaves) {
        const int m0 = tile << 4;
        BfPack A[4];
        const unsigned gOff = (unsigned)(m0 + r) * 64u + (unsigned)half * 8u;
#pragma unroll
        for (int c = 0; c < 2; ++c) {                 // K 0..63 : agg (fp32->bf16)
#pragma unroll
            for (int i = 0; i < 8; ++i) {
                A[c].s[i]     = to_bf16_bits(agg[gOff + (unsigned)(c * 32 + i)]);
                A[c].s[i + 8] = to_bf16_bits(agg[gOff + (unsigned)(c * 32 + 16 + i)]);
            }
        }
#pragma unroll
        for (int c = 0; c < 2; ++c) {                 // K 64..127 : h_n (bf16)
            A[2 + c].us2[0] = *(const v8us*)(h_n_bf + gOff + (unsigned)(c * 32));
            A[2 + c].us2[1] = *(const v8us*)(h_n_bf + gOff + (unsigned)(c * 32 + 16));
        }
        // out base: element (i,t) lands at base + i*96 + t*16
        const unsigned oOff = (unsigned)(m0 + 8 * half) * 96u + 32u + (unsigned)r;
#pragma unroll
        for (int t = 0; t < 4; ++t) {
            v8f c = {};
#pragma unroll
            for (int cc = 0; cc < 4; ++cc) {
                v16bf b = load_B_lds(sBn, cc, t, lane);
                c = __builtin_amdgcn_wmma_f32_16x16x32_bf16(false, A[cc].bf, false, b,
                                                            (short)0, c, false, false);
            }
#pragma unroll
            for (int i = 0; i < 8; ++i) {
                float y = swishf(c[i] + bv[t]);
                out[oOff + (unsigned)(i * 96 + t * 16)] = y;
            }
        }
        // copy nodes0 -> out[:, 0:32]  (lanes cover rows x 32 cols, coalesced)
        const unsigned srcBase = (unsigned)m0 * 32u;
        const unsigned dstBase = (unsigned)m0 * 96u;
#pragma unroll
        for (int jj = 0; jj < 16; ++jj) {
            int j   = jj * 32 + lane;
            int mm  = j >> 5;
            int col = j & 31;
            out[dstBase + (unsigned)(mm * 96 + col)] = nodes[srcBase + (unsigned)(mm * 32 + col)];
        }
    }
}

// ---------------------------------------------------------------------------
extern "C" void kernel_launch(void* const* d_in, const int* in_sizes, int n_in,
                              void* d_out, int out_size, void* d_ws, size_t ws_size,
                              hipStream_t stream) {
    const float* nodes     = (const float*)d_in[0];
    const float* edges     = (const float*)d_in[1];
    const int*   senders   = (const int*)  d_in[2];
    const int*   receivers = (const int*)  d_in[3];
    const float* W_emb_n   = (const float*)d_in[4];
    const float* b_emb_n   = (const float*)d_in[5];
    const float* W_emb_e   = (const float*)d_in[6];
    const float* b_emb_e   = (const float*)d_in[7];
    const float* W_e       = (const float*)d_in[8];
    const float* b_e       = (const float*)d_in[9];
    const float* W_n       = (const float*)d_in[10];
    const float* b_n       = (const float*)d_in[11];
    float* out = (float*)d_out;

    const int N = in_sizes[0] / 32;   // 50000
    const int E = in_sizes[2];        // 1000000

    // workspace: h_n bf16 [N,64], then agg f32 [N,64]
    unsigned short* h_n_bf = (unsigned short*)d_ws;
    size_t hbytes = (((size_t)N * 64 * 2) + 255) & ~(size_t)255;
    float* agg = (float*)((char*)d_ws + hbytes);

    const int zTot = N * 64;
    k_zero<<<dim3((zTot + 255) / 256), dim3(256), 0, stream>>>(agg, zTot);

    const int nTilesN = N / 16;
    const int nTilesE = E / 16;
    int blkN = (nTilesN + WAVES_PER_BLOCK - 1) / WAVES_PER_BLOCK;
    if (blkN > 1024) blkN = 1024;
    if (blkN < 1)    blkN = 1;
    int blkE = (nTilesE + WAVES_PER_BLOCK - 1) / WAVES_PER_BLOCK;
    if (blkE > 1024) blkE = 1024;
    if (blkE < 1)    blkE = 1;

    k_node_embed<<<dim3(blkN), dim3(256), 0, stream>>>(nodes, W_emb_n, b_emb_n, h_n_bf, N);
    k_edge<<<dim3(blkE), dim3(256), 0, stream>>>(edges, senders, receivers,
                                                 W_emb_e, b_emb_e, W_e, b_e,
                                                 h_n_bf, agg, E);
    k_node<<<dim3(blkN), dim3(256), 0, stream>>>(nodes, W_n, b_n, agg, h_n_bf, out, N);
}
